// BatchFunctionalCNN_63273458205360
// MI455X (gfx1250) — matrix operand order repaired and checked by reading, not compiled, not run on hardware
//
#include <hip/hip_runtime.h>
#include <hip/hip_bf16.h>
#include <cstdint>

// =====================================================================
// Batched functional CNN for MI455X (gfx1250), bf16 WMMA implicit GEMM.
// All matrix math via V_WMMA_F32_16X16X32_BF16 (fp32 accumulate).
// Activations ping-pong through d_ws in bf16 channel-last layout; each
// layer's tensors (<=64MB) stay L2-resident (192MB L2).
// Wave id is readfirstlane'd so tile loops are scalar (uniform) loops
// and EXEC is provably all-ones at every WMMA (ISA 7.12 requirement).
// LDS weight staging is B128-vectorized; next-tile activations are
// prefetched (global_prefetch_b8) under the current tile's WMMAs.
// =====================================================================

typedef __attribute__((ext_vector_type(16))) __bf16 v16bf;
typedef __attribute__((ext_vector_type(8)))  float  v8f;

union BF8U { uint4 u; __bf16 h[8]; };

__device__ __forceinline__ v16bf frag_zero() {
  v16bf f;
#pragma unroll
  for (int e = 0; e < 16; ++e) f[e] = (__bf16)0.0f;
  return f;
}

// Build a 16-element bf16 fragment from two 16-byte chunks.
// Elements [0..7] <- p0[0..7], elements [8..15] <- p1[0..7].
__device__ __forceinline__ v16bf frag_pair(const __bf16* p0, const __bf16* p1) {
  BF8U a, b;
  a.u = *reinterpret_cast<const uint4*>(p0);
  b.u = *reinterpret_cast<const uint4*>(p1);
  v16bf f;
#pragma unroll
  for (int e = 0; e < 8; ++e) { f[e] = a.h[e]; f[8 + e] = b.h[e]; }
  return f;
}

__device__ __forceinline__ int wave_id() {
  return __builtin_amdgcn_readfirstlane((int)(threadIdx.x >> 5));
}

// B128-vectorized global->LDS copy (nelem bf16, both 16B aligned,
// nelem % 8 == 0). 256 threads.
__device__ __forceinline__ void stage_lds(__bf16* dst, const __bf16* src,
                                          int nelem, int tid) {
  uint4* d = reinterpret_cast<uint4*>(dst);
  const uint4* s = reinterpret_cast<const uint4*>(src);
  const int n4 = nelem >> 3;
  for (int i = tid; i < n4; i += 256) d[i] = s[i];
}

// ----------------------------- prep kernels --------------------------

// x [128][3][32][32] f32 -> xbf [img][y][x][ci] bf16
__global__ __launch_bounds__(256) void prep_x(const float* __restrict__ x,
                                              __bf16* __restrict__ xbf, int n) {
  int i = blockIdx.x * 256 + threadIdx.x;
  if (i >= n) return;
  int ci = i % 3; int p = i / 3;
  int xx = p & 31; p >>= 5;
  int yy = p & 31; int img = p >> 5;
  xbf[i] = (__bf16)x[(((size_t)img * 3 + ci) * 32 + yy) * 32 + xx];
}

// w0 (8,3,32,9) -> wp0[mb][co][K32], K = tap*3 + ci, zero-padded 27..31
__global__ __launch_bounds__(256) void prep_w0(const float* __restrict__ w0,
                                               __bf16* __restrict__ wp, int n) {
  int i = blockIdx.x * 256 + threadIdx.x;
  if (i >= n) return;
  int k = i & 31; int p = i >> 5;
  int co = p & 31; int mb = p >> 5;
  float v = 0.f;
  if (k < 27) { int t = k / 3, ci = k - t * 3;
                v = w0[(((size_t)mb * 3 + ci) * 32 + co) * 9 + t]; }
  wp[i] = (__bf16)v;
}

// w_l (8,32,32,9) -> wp[mb][tap][co][ci]
__global__ __launch_bounds__(256) void prep_wc(const float* __restrict__ w,
                                               __bf16* __restrict__ wp, int n) {
  int i = blockIdx.x * 256 + threadIdx.x;
  if (i >= n) return;
  int ci = i & 31; int p = i >> 5;
  int co = p & 31; p >>= 5;
  int t = p % 9; int mb = p / 9;
  wp[i] = (__bf16)w[(((size_t)mb * 32 + ci) * 32 + co) * 9 + t];
}

// w7 (8,32,256,16) -> wp7[mb][o][k], k = hw*32 + ci (memory order of our acts)
__global__ __launch_bounds__(256) void prep_w7(const float* __restrict__ w7,
                                               __bf16* __restrict__ wp, int n) {
  int i = blockIdx.x * 256 + threadIdx.x;
  if (i >= n) return;
  int k = i & 511; int p = i >> 9;
  int o = p & 255; int mb = p >> 8;
  int hw = k >> 5; int ci = k & 31;
  wp[i] = (__bf16)w7[(((size_t)mb * 32 + ci) * 256 + o) * 16 + hw];
}

// w8 (8,256,10,1) -> wp8[mb][oo(16, zero-padded)][cin(256)]
__global__ __launch_bounds__(256) void prep_w8(const float* __restrict__ w8,
                                               __bf16* __restrict__ wp, int n) {
  int i = blockIdx.x * 256 + threadIdx.x;
  if (i >= n) return;
  int cin = i & 255; int p = i >> 8;
  int oo = p & 15; int mb = p >> 4;
  float v = (oo < 10) ? w8[((size_t)mb * 256 + cin) * 10 + oo] : 0.f;
  wp[i] = (__bf16)v;
}

// ----------------------------- conv0 ---------------------------------
// One block per (mb,img). K = 27 (9 taps x 3 cin) padded to 32. Each
// wave handles M-tiles; one A gather feeds both N-tiles (2 WMMAs).
__global__ __launch_bounds__(256) void conv0_wmma(
    const __bf16* __restrict__ xbf, const __bf16* __restrict__ wp0,
    const float* __restrict__ b0, __bf16* __restrict__ out) {
  __shared__ __align__(16) __bf16 ilds[32 * 32 * 3];
  __shared__ __align__(16) __bf16 wlds[32 * 32];
  __shared__ float blds[32];
  const int tid = threadIdx.x;
  const int blk = blockIdx.x;       // inst = mb*128 + img
  const int mb = blk >> 7;
  const int img = blk & 127;
  stage_lds(ilds, xbf + (size_t)img * 3072, 3072, tid);
  stage_lds(wlds, wp0 + (size_t)mb * 1024, 1024, tid);
  if (tid < 32) blds[tid] = b0[mb * 32 + tid];
  __syncthreads();

  const int wid = wave_id();
  const int lane = tid & 31;
  const int m = lane & 15;
  const int klo = (lane < 16) ? 0 : 16;   // A-frag K base (contiguous, matches B)
  const int kb2 = (lane < 16) ? 0 : 16;   // B-frag K base

  const __bf16* row0 = &wlds[m * 32 + kb2];
  const __bf16* row1 = &wlds[(16 + m) * 32 + kb2];
  const v16bf b0f = frag_pair(row0, row0 + 8);
  const v16bf b1f = frag_pair(row1, row1 + 8);
  const float bv0 = blds[m], bv1 = blds[16 + m];

  for (int mtile = wid; mtile < 64; mtile += 8) {
    const int mbase = mtile * 16;
    const int pix = mbase + m;
    const int y = pix >> 5, x = pix & 31;
    v16bf a = frag_zero();
#pragma unroll
    for (int e = 0; e < 16; ++e) {
      const int K = klo + e;
      if (K < 27) {
        const int t = K / 3, ci = K - t * 3;
        const int yy = y + t / 3 - 1, xx = x + t % 3 - 1;
        if (yy >= 0 && yy < 32 && xx >= 0 && xx < 32)
          a[e] = ilds[(yy * 32 + xx) * 3 + ci];
      }
    }
    v8f acc0 = {}, acc1 = {};
    acc0 = __builtin_amdgcn_wmma_f32_16x16x32_bf16(false, a, false, b0f,
                                                   (short)0, acc0, false, false);
    acc1 = __builtin_amdgcn_wmma_f32_16x16x32_bf16(false, a, false, b1f,
                                                   (short)0, acc1, false, false);
    const int prow = mbase + ((lane >> 4) << 3);
#pragma unroll
    for (int r = 0; r < 8; ++r) {
      float v0 = acc0[r] + bv0; v0 = v0 > 0.f ? v0 : 0.f;
      float v1 = acc1[r] + bv1; v1 = v1 > 0.f ? v1 : 0.f;
      out[(size_t)blk * 32768 + (size_t)(prow + r) * 32 + m]      = (__bf16)v0;
      out[(size_t)blk * 32768 + (size_t)(prow + r) * 32 + 16 + m] = (__bf16)v1;
    }
  }
}

// ----------------------------- conv1..6 ------------------------------
// SAME 3x3 conv 32->32 on SxS (S compile-time), implicit GEMM. Per
// M-tile: 9 taps, each = one A load (2x16B) + 2 B LDS loads + 2 WMMAs
// (both output-channel halves). Weights (18KB) staged in LDS via B128.
template <int S>
__global__ __launch_bounds__(256) void conv3x3_wmma(
    const __bf16* __restrict__ in, __bf16* __restrict__ out,
    const __bf16* __restrict__ wp, const float* __restrict__ bias) {
  __shared__ __align__(16) __bf16 wlds[9 * 32 * 32];
  __shared__ float blds[32];
  const int tid = threadIdx.x;
  const int blk = blockIdx.x;
  const int mb = blk >> 7;
  stage_lds(wlds, wp + (size_t)mb * 9216, 9216, tid);
  if (tid < 32) blds[tid] = bias[mb * 32 + tid];
  __syncthreads();

  const int wid = wave_id();
  const int lane = tid & 31;
  const int m = lane & 15;
  const int klo = (lane < 16) ? 0 : 8;
  const int kb2 = (lane < 16) ? 0 : 16;
  constexpr int SHIFT = (S == 32) ? 5 : (S == 16) ? 4 : 3;
  constexpr int MASK = S - 1;
  constexpr int stride = S * S * 32;
  constexpr int nM = S * S / 16;
  const __bf16* ibase = in + (size_t)blk * stride;
  const float bv0 = blds[m], bv1 = blds[16 + m];

  for (int mtile = wid; mtile < nM; mtile += 8) {
    const int mbase = mtile * 16;
    const int pix = mbase + m;
    const int y = pix >> SHIFT, x = pix & MASK;

    // Prefetch next M-tile's center row of activations into cache
    // while this tile's 18 WMMAs run (speculative, no counter cost).
    if (mtile + 8 < nM) {
      const int npix = (mtile + 8) * 16 + m;
      __builtin_prefetch(ibase + (size_t)npix * 32, 0, 1);
    }

    v8f acc0 = {}, acc1 = {};
#pragma unroll
    for (int t = 0; t < 9; ++t) {
      const int yy = y + t / 3 - 1;
      const int xx = x + t % 3 - 1;
      v16bf a;
      if (yy >= 0 && yy < S && xx >= 0 && xx < S) {
        const __bf16* p = ibase + (size_t)((yy << SHIFT) + xx) * 32;
        a = frag_pair(p + kb2, p + kb2 + 8);
      } else {
        a = frag_zero();   // SAME zero padding
      }
      const __bf16* row0 = &wlds[(t * 32 + m) * 32 + kb2];
      const __bf16* row1 = &wlds[(t * 32 + 16 + m) * 32 + kb2];
      v16bf bf0 = frag_pair(row0, row0 + 8);
      v16bf bf1 = frag_pair(row1, row1 + 8);
      acc0 = __builtin_amdgcn_wmma_f32_16x16x32_bf16(false, a, false, bf0,
                                                     (short)0, acc0, false, false);
      acc1 = __builtin_amdgcn_wmma_f32_16x16x32_bf16(false, a, false, bf1,
                                                     (short)0, acc1, false, false);
    }
    const int prow = mbase + ((lane >> 4) << 3);
#pragma unroll
    for (int r = 0; r < 8; ++r) {
      float v0 = acc0[r] + bv0; v0 = v0 > 0.f ? v0 : 0.f;  // ReLU fused
      float v1 = acc1[r] + bv1; v1 = v1 > 0.f ? v1 : 0.f;
      out[(size_t)blk * stride + (size_t)(prow + r) * 32 + m]      = (__bf16)v0;
      out[(size_t)blk * stride + (size_t)(prow + r) * 32 + 16 + m] = (__bf16)v1;
    }
  }
}

// ----------------------------- maxpool 2x2 ---------------------------
__global__ __launch_bounds__(256) void maxpool2_kernel(
    const __bf16* __restrict__ in, __bf16* __restrict__ out, int Sin, int total) {
  int tid = blockIdx.x * 256 + threadIdx.x;
  if (tid >= total) return;
  const int So = Sin >> 1;
  const int c = tid & 31;
  int p = tid >> 5;
  const int ox = p % So; p /= So;
  const int oy = p % So; p /= So;
  const int inst = p;
  const __bf16* ib = in + (size_t)inst * Sin * Sin * 32;
  float v0 = (float)ib[((2 * oy) * Sin + 2 * ox) * 32 + c];
  float v1 = (float)ib[((2 * oy) * Sin + 2 * ox + 1) * 32 + c];
  float v2 = (float)ib[((2 * oy + 1) * Sin + 2 * ox) * 32 + c];
  float v3 = (float)ib[((2 * oy + 1) * Sin + 2 * ox + 1) * 32 + c];
  float mx = fmaxf(fmaxf(v0, v1), fmaxf(v2, v3));
  out[(size_t)inst * So * So * 32 + (oy * So + ox) * 32 + c] = (__bf16)mx;
}

// ----------------------------- fc7 -----------------------------------
// out[mb][n][o] = relu( sum_k in[mb,n][k] * w7p[mb][o][k] + b7 ), K=512.
__global__ __launch_bounds__(256) void fc7_wmma(
    const __bf16* __restrict__ in,    // [1024][512] pooled+relu'd conv6
    const __bf16* __restrict__ wp7,   // [8][256][512]
    const float* __restrict__ b7,     // [8][256]
    __bf16* __restrict__ out) {       // [8][128][256]
  __shared__ float blds[256];
  const int tid = threadIdx.x;
  const int mb = blockIdx.x;
  blds[tid] = b7[mb * 256 + tid];
  __syncthreads();
  const int wid = wave_id();
  const int lane = tid & 31;
  const int m = lane & 15;
  const int klo = (lane < 16) ? 0 : 8;
  const int kb2 = (lane < 16) ? 0 : 16;
  for (int tile = wid; tile < 128; tile += 8) {
    const int mbase = (tile >> 4) * 16;
    const int nbase = (tile & 15) * 16;
    const __bf16* abase = in + ((size_t)(mb * 128 + mbase + m)) * 512;
    const __bf16* bbase = wp7 + ((size_t)(mb * 256 + nbase + m)) * 512;
    v8f acc = {};
#pragma unroll
    for (int kb = 0; kb < 16; ++kb) {
      v16bf a = frag_pair(abase + kb * 32 + kb2, abase + kb * 32 + kb2 + 8);
      v16bf b = frag_pair(bbase + kb * 32 + kb2, bbase + kb * 32 + kb2 + 8);
      acc = __builtin_amdgcn_wmma_f32_16x16x32_bf16(false, a, false, b,
                                                    (short)0, acc, false, false);
    }
    const float bv = blds[nbase + m];
    const int irow = mbase + ((lane >> 4) << 3);
#pragma unroll
    for (int r = 0; r < 8; ++r) {
      float v = acc[r] + bv;
      v = v > 0.f ? v : 0.f;  // ReLU (consumed by fc8)
      out[((size_t)(mb * 128 + irow + r)) * 256 + nbase + m] = (__bf16)v;
    }
  }
}

// ----------------------------- fc8 -----------------------------------
// out[mb][n][oo] = sum_k relu(fc7)[n][k] * w8p[mb][oo][k] + b8, oo<10, f32.
__global__ __launch_bounds__(256) void fc8_wmma(
    const __bf16* __restrict__ in,    // [8][128][256]
    const __bf16* __restrict__ wp8,   // [8][16][256]
    const float* __restrict__ b8,     // [8][10]
    float* __restrict__ out) {        // [8][128][10]
  __shared__ float blds[16];
  const int tid = threadIdx.x;
  const int mb = blockIdx.x;
  if (tid < 16) blds[tid] = (tid < 10) ? b8[mb * 10 + tid] : 0.f;
  __syncthreads();
  const int wid = wave_id();
  const int lane = tid & 31;
  const int m = lane & 15;
  const int klo = (lane < 16) ? 0 : 8;
  const int kb2 = (lane < 16) ? 0 : 16;
  const int mbase = wid * 16;   // 8 waves cover all 8 M-tiles
  const __bf16* abase = in + ((size_t)(mb * 128 + mbase + m)) * 256;
  const __bf16* bbase = wp8 + ((size_t)(mb * 16 + m)) * 256;
  v8f acc = {};
#pragma unroll
  for (int kb = 0; kb < 8; ++kb) {
    v16bf a = frag_pair(abase + kb * 32 + kb2, abase + kb * 32 + kb2 + 8);
    v16bf b = frag_pair(bbase + kb * 32 + kb2, bbase + kb * 32 + kb2 + 8);
    acc = __builtin_amdgcn_wmma_f32_16x16x32_bf16(false, a, false, b,
                                                  (short)0, acc, false, false);
  }
  const float bv = blds[m];
  const int irow = mbase + ((lane >> 4) << 3);
  if (m < 10) {
#pragma unroll
    for (int r = 0; r < 8; ++r)
      out[((size_t)(mb * 128 + irow + r)) * 10 + m] = acc[r] + bv;
  }
}

// ----------------------------- launch --------------------------------
extern "C" void kernel_launch(void* const* d_in, const int* in_sizes, int n_in,
                              void* d_out, int out_size, void* d_ws, size_t ws_size,
                              hipStream_t stream) {
  (void)in_sizes; (void)n_in; (void)out_size;
  // dict order: x, w0,b0, w1,b1, ..., w8,b8
  const float* x = (const float*)d_in[0];
  const float* w[9]; const float* b[9];
  for (int i = 0; i < 9; ++i) {
    w[i] = (const float*)d_in[1 + 2 * i];
    b[i] = (const float*)d_in[2 + 2 * i];
  }

  char* p = (char*)d_ws;
  auto carve = [&](size_t elems) {
    __bf16* r = (__bf16*)p; p += elems * sizeof(__bf16); return r;
  };
  __bf16* XBF  = carve(393216);    // x bf16, channel-last
  __bf16* WP0  = carve(8192);      // conv0 weights [mb][co][K32]
  __bf16* WPC  = carve(442368);    // conv1..6 weights [l][mb][t][co][ci]
  __bf16* WP7  = carve(1048576);   // fc7 weights [mb][o][512]
  __bf16* WP8  = carve(32768);     // fc8 weights [mb][16][256]
  __bf16* FC7O = carve(262144);    // fc7 activations [mb][128][256]
  __bf16* BUFA = carve(33554432);  // activation ping (64MB, L2-resident)
  __bf16* BUFB = carve(33554432);  // activation pong
  if ((size_t)(p - (char*)d_ws) > ws_size) return;  // need ~139MB scratch

  auto nb = [](int n) { return (n + 255) / 256; };
  prep_x<<<nb(393216), 256, 0, stream>>>(x, XBF, 393216);
  prep_w0<<<nb(8192), 256, 0, stream>>>(w[0], WP0, 8192);
  for (int l = 1; l <= 6; ++l)
    prep_wc<<<nb(73728), 256, 0, stream>>>(w[l], WPC + (size_t)(l - 1) * 73728, 73728);
  prep_w7<<<nb(1048576), 256, 0, stream>>>(w[7], WP7, 1048576);
  prep_w8<<<nb(32768), 256, 0, stream>>>(w[8], WP8, 32768);

  conv0_wmma<<<1024, 256, 0, stream>>>(XBF, WP0, b[0], BUFA);
  conv3x3_wmma<32><<<1024, 256, 0, stream>>>(BUFA, BUFB, WPC + 0 * 73728, b[1]);
  conv3x3_wmma<32><<<1024, 256, 0, stream>>>(BUFB, BUFA, WPC + 1 * 73728, b[2]);
  maxpool2_kernel<<<nb(1024 * 16 * 16 * 32), 256, 0, stream>>>(BUFA, BUFB, 32, 1024 * 16 * 16 * 32);
  conv3x3_wmma<16><<<1024, 256, 0, stream>>>(BUFB, BUFA, WPC + 2 * 73728, b[3]);
  conv3x3_wmma<16><<<1024, 256, 0, stream>>>(BUFA, BUFB, WPC + 3 * 73728, b[4]);
  maxpool2_kernel<<<nb(1024 * 8 * 8 * 32), 256, 0, stream>>>(BUFB, BUFA, 16, 1024 * 8 * 8 * 32);
  conv3x3_wmma<8><<<1024, 256, 0, stream>>>(BUFA, BUFB, WPC + 4 * 73728, b[5]);
  conv3x3_wmma<8><<<1024, 256, 0, stream>>>(BUFB, BUFA, WPC + 5 * 73728, b[6]);
  maxpool2_kernel<<<nb(1024 * 4 * 4 * 32), 256, 0, stream>>>(BUFA, BUFB, 8, 1024 * 4 * 4 * 32);
  fc7_wmma<<<8, 256, 0, stream>>>(BUFB, WP7, b[7], FC7O);
  fc8_wmma<<<8, 256, 0, stream>>>(FC7O, WP8, b[8], (float*)d_out);
}
